// ContextModule_24739011625870
// MI455X (gfx1250) — matrix operand order repaired and checked by reading, not compiled, not run on hardware
//
#include <hip/hip_runtime.h>
#include <hip/hip_bf16.h>

// ---------------------------------------------------------------------------
// Problem dims (compile-time constants from the reference)
// ---------------------------------------------------------------------------
#define DIM_IN   2048
#define DIM_CTX  512
#define DIM_KEY  128
#define DIM_VAL  128
#define NCLIP    8
#define S_TOT    16384   // T*H*W = 16*32*32
#define NBOX     512

typedef __attribute__((ext_vector_type(16))) __bf16 v16bf;
typedef __attribute__((ext_vector_type(8)))  float  v8f;

// ---------------------------------------------------------------------------
// WMMA wrapper: D = A(16x32 bf16) * B(32x16 bf16) + C(16x16 f32)
// ---------------------------------------------------------------------------
static __device__ inline v8f wmma_bf16(v16bf a, v16bf b, v8f c) {
  return __builtin_amdgcn_wmma_f32_16x16x32_bf16(
      /*neg_a=*/false, a, /*neg_b=*/false, b,
      /*c_mod=*/(short)0, c, /*reuse_a=*/false, /*reuse_b=*/false);
}

// ---------------------------------------------------------------------------
// A fragment (16x32, bf16): lane m = lane&15, half = lane>>4.
// VGPR v holds K pair { (v>>2)*16 + half*8 + (v&3)*2, +1 } (ISA 7.12.2 table).
// ---------------------------------------------------------------------------
static __device__ inline v16bf afrag_f32(const float* A, int lda, int m0, int k0, int lane) {
  int m = m0 + (lane & 15);
  int half = lane >> 4;
  const float* base = A + (size_t)m * lda + k0 + half * 8;
  v16bf a;
#pragma unroll
  for (int v = 0; v < 8; ++v) {
    const float* p = base + (v >> 2) * 16 + (v & 3) * 2;
    a[2 * v]     = (__bf16)p[0];
    a[2 * v + 1] = (__bf16)p[1];
  }
  return a;
}

static __device__ inline v16bf afrag_bf16(const __bf16* A, int lda, int m0, int k0, int lane) {
  int m = m0 + (lane & 15);
  int half = lane >> 4;
  const __bf16* base = A + (size_t)m * lda + k0 + half * 8;
  v16bf a;
#pragma unroll
  for (int v = 0; v < 8; ++v) {
    const __bf16* p = base + (v >> 2) * 16 + (v & 3) * 2;
    a[2 * v]     = p[0];
    a[2 * v + 1] = p[1];
  }
  return a;
}

// ---------------------------------------------------------------------------
// B fragment (32x16, bf16): lane n = n0+(lane&15), half = lane>>4; element j
// holds K = (j>>3)*16 + half*8 + (j&7). Generic strides: element(k,n) = P[k*sk + n*sn].
// ---------------------------------------------------------------------------
static __device__ inline v16bf bfrag_f32s(const float* B, size_t sk, size_t sn,
                                          int k0, int n0, int lane) {
  const float* base = B + (size_t)(k0 + (lane >> 4) * 8) * sk + (size_t)(n0 + (lane & 15)) * sn;
  v16bf b;
#pragma unroll
  for (int j = 0; j < 16; ++j) b[j] = (__bf16)base[(size_t)((j >> 3) * 16 + (j & 7)) * sk];
  return b;
}

static __device__ inline v16bf bfrag_bf16s(const __bf16* B, size_t sk, size_t sn,
                                           int k0, int n0, int lane) {
  const __bf16* base = B + (size_t)(k0 + (lane >> 4) * 8) * sk + (size_t)(n0 + (lane & 15)) * sn;
  v16bf b;
#pragma unroll
  for (int j = 0; j < 16; ++j) b[j] = base[(size_t)((j >> 3) * 16 + (j & 7)) * sk];
  return b;
}

// ---------------------------------------------------------------------------
// Kernel 1: q = rois @ W_proj^T + b_proj, folded attention scale, bf16 out.
// M=512 boxes, N=128 key dims, K=2048.  32x8 = 256 tiles = 32 blocks x 8 waves.
// ---------------------------------------------------------------------------
__global__ void qproj_kernel(const float* __restrict__ rois,
                             const float* __restrict__ Wp,
                             const float* __restrict__ bp,
                             __bf16* __restrict__ qbf) {
  int lane = threadIdx.x & 31;
  int wg = blockIdx.x * 8 + (threadIdx.x >> 5);
  int nt = wg & 7;
  int mt = wg >> 3;
  int m0 = mt * 16, n0 = nt * 16;
  v8f acc = {};
  for (int k0 = 0; k0 < DIM_IN; k0 += 32) {
    v16bf a = afrag_f32(rois, DIM_IN, m0, k0, lane);
    // B(k,n) = W_proj[n][k]  (W_proj is [DIM_KEY, DIM_IN] row-major)
    v16bf b = bfrag_f32s(Wp, (size_t)1, (size_t)DIM_IN, k0, n0, lane);
    acc = wmma_bf16(a, b, acc);
  }
  int cn = n0 + (lane & 15);
  int half = lane >> 4;
  const float scale = 0.08838834764831845f; // 1/sqrt(128)
  float bias = bp[cn];
#pragma unroll
  for (int v = 0; v < 8; ++v) {
    int cm = m0 + v + 8 * half;
    qbf[(size_t)cm * DIM_KEY + cn] = (__bf16)((acc[v] + bias) * scale);
  }
}

// ---------------------------------------------------------------------------
// Kernel 2: fused keys/vals 1x1x1 conv: per clip n,
//   keysB[n][k][s] = W_keys@ctx + b_keys   (bf16, [128 x S])
//   valsB[n][v][s] = W_vals@ctx + b_vals   (bf16, [128 x S], transposed vals)
// Two WMMAs per K-step share one ctx B-fragment.
// Waves: 8 clips * 8 m-tiles * 1024 s-tiles = 65536 -> 8192 blocks.
// ---------------------------------------------------------------------------
__global__ void kv_proj_kernel(const float* __restrict__ ctx,
                               const float* __restrict__ Wk,
                               const float* __restrict__ bk,
                               const float* __restrict__ Wv,
                               const float* __restrict__ bv,
                               __bf16* __restrict__ keysB,
                               __bf16* __restrict__ valsB) {
  int lane = threadIdx.x & 31;
  int wg = blockIdx.x * 8 + (threadIdx.x >> 5);
  int st = wg & 1023;
  int mt = (wg >> 10) & 7;
  int n  = wg >> 13;
  int m0 = mt * 16, s0 = st * 16;
  int half = lane >> 4;

  v8f acck, accv;
#pragma unroll
  for (int v = 0; v < 8; ++v) {
    int cm = m0 + v + 8 * half;
    acck[v] = bk[cm];
    accv[v] = bv[cm];
  }
  const float* ctxn = ctx + (size_t)n * DIM_CTX * S_TOT;
  for (int c0 = 0; c0 < DIM_CTX; c0 += 32) {
    v16bf bb = bfrag_f32s(ctxn, (size_t)S_TOT, (size_t)1, c0, s0, lane);
    v16bf ak = afrag_f32(Wk, DIM_CTX, m0, c0, lane);
    v16bf av = afrag_f32(Wv, DIM_CTX, m0, c0, lane);
    acck = wmma_bf16(ak, bb, acck);
    accv = wmma_bf16(av, bb, accv);
  }
  int cs = s0 + (lane & 15);
  size_t base = (size_t)n * DIM_KEY * S_TOT;
#pragma unroll
  for (int v = 0; v < 8; ++v) {
    int cm = m0 + v + 8 * half;
    keysB[base + (size_t)cm * S_TOT + cs] = (__bf16)acck[v];
    valsB[base + (size_t)cm * S_TOT + cs] = (__bf16)accv[v];
  }
}

// ---------------------------------------------------------------------------
// Kernel 3: qk[b][s] = q[b] . keys[clip][: ][s]  (scale pre-folded into q).
// Grid: clip n (8) x box-tile (32) x s-tile (1024) -> 32768 blocks x 8 waves.
// batch_idx is sorted -> wave-uniform early exit keeps EXEC all-1s for WMMA.
// Rows are stored only when batch_idx[row]==n (boundary tiles run twice).
// Output written straight into d_out's attn region (f32).
// ---------------------------------------------------------------------------
__global__ void qk_kernel(const __bf16* __restrict__ qbf,
                          const __bf16* __restrict__ keysB,
                          const int* __restrict__ bidx,
                          float* __restrict__ qk_out) {
  int lane = threadIdx.x & 31;
  int wg = blockIdx.x * 8 + (threadIdx.x >> 5);
  int st = wg & 1023;
  int bt = (wg >> 10) & 31;
  int n  = wg >> 15;
  int b0 = bt * 16;
  if (n < bidx[b0] || n > bidx[b0 + 15]) return;  // wave-uniform
  int s0 = st * 16;
  v8f acc = {};
  const __bf16* K = keysB + (size_t)n * DIM_KEY * S_TOT;
#pragma unroll
  for (int k0 = 0; k0 < DIM_KEY; k0 += 32) {
    v16bf a = afrag_bf16(qbf, DIM_KEY, b0, k0, lane);
    v16bf b = bfrag_bf16s(K, (size_t)S_TOT, (size_t)1, k0, s0, lane);
    acc = wmma_bf16(a, b, acc);
  }
  int cs = s0 + (lane & 15);
  int half = lane >> 4;
#pragma unroll
  for (int v = 0; v < 8; ++v) {
    int cb = b0 + v + 8 * half;
    if (bidx[cb] == n) qk_out[(size_t)cb * S_TOT + cs] = acc[v];
  }
}

// ---------------------------------------------------------------------------
// Kernel 4: per-box softmax over S. In-place: attn row becomes coeff*255
// (final output); bf16 coeff copy stored for the AV GEMM.
// ---------------------------------------------------------------------------
__global__ void softmax_kernel(float* __restrict__ attn,
                               __bf16* __restrict__ coeffB) {
  int b = blockIdx.x;
  int tid = threadIdx.x;
  float* row = attn + (size_t)b * S_TOT;
  __shared__ float red[256];

  float m = -3.402823466e+38f;
  for (int i = tid; i < S_TOT; i += 256) m = fmaxf(m, row[i]);
  red[tid] = m; __syncthreads();
  for (int s = 128; s > 0; s >>= 1) {
    if (tid < s) red[tid] = fmaxf(red[tid], red[tid + s]);
    __syncthreads();
  }
  m = red[0]; __syncthreads();

  float sum = 0.f;
  for (int i = tid; i < S_TOT; i += 256) sum += __expf(row[i] - m);
  red[tid] = sum; __syncthreads();
  for (int s = 128; s > 0; s >>= 1) {
    if (tid < s) red[tid] += red[tid + s];
    __syncthreads();
  }
  float inv = 1.0f / red[0];

  for (int i = tid; i < S_TOT; i += 256) {
    float c = __expf(row[i] - m) * inv;
    coeffB[(size_t)b * S_TOT + i] = (__bf16)c;
    row[i] = c * 255.0f;
  }
}

// ---------------------------------------------------------------------------
// Kernel 5: ctx_out[b][v] = sum_s coeff[b][s] * valsT[clip][v][s].
// Grid: clip (8) x box-tile (32) x v-tile (8) -> 256 blocks x 8 waves;
// K loop over S (512 WMMA steps). Same clip-masking trick as qk.
// ---------------------------------------------------------------------------
__global__ void av_kernel(const __bf16* __restrict__ coeffB,
                          const __bf16* __restrict__ valsB,
                          const int* __restrict__ bidx,
                          float* __restrict__ ctx_out) {
  int lane = threadIdx.x & 31;
  int wg = blockIdx.x * 8 + (threadIdx.x >> 5);
  int vt = wg & 7;
  int bt = (wg >> 3) & 31;
  int n  = wg >> 8;
  int b0 = bt * 16;
  if (n < bidx[b0] || n > bidx[b0 + 15]) return;  // wave-uniform
  int v0 = vt * 16;
  v8f acc = {};
  const __bf16* V = valsB + (size_t)n * DIM_VAL * S_TOT;
  for (int s0 = 0; s0 < S_TOT; s0 += 32) {
    v16bf a = afrag_bf16(coeffB, S_TOT, b0, s0, lane);
    // B(k=s, n=v) = valsT[v][s] -> sk=1, sn=S_TOT
    v16bf b = bfrag_bf16s(V, (size_t)1, (size_t)S_TOT, s0, v0, lane);
    acc = wmma_bf16(a, b, acc);
  }
  int cv = v0 + (lane & 15);
  int half = lane >> 4;
#pragma unroll
  for (int v = 0; v < 8; ++v) {
    int cb = b0 + v + 8 * half;
    if (bidx[cb] == n) ctx_out[(size_t)cb * DIM_VAL + cv] = acc[v];
  }
}

// ---------------------------------------------------------------------------
// Launcher
// ---------------------------------------------------------------------------
extern "C" void kernel_launch(void* const* d_in, const int* in_sizes, int n_in,
                              void* d_out, int out_size, void* d_ws, size_t ws_size,
                              hipStream_t stream) {
  const float* rois   = (const float*)d_in[0];
  const float* ctx    = (const float*)d_in[1];
  const int*   bidx   = (const int*)  d_in[2];
  const float* W_proj = (const float*)d_in[3];
  const float* b_proj = (const float*)d_in[4];
  const float* W_keys = (const float*)d_in[5];
  const float* b_keys = (const float*)d_in[6];
  const float* W_vals = (const float*)d_in[7];
  const float* b_vals = (const float*)d_in[8];

  float* ctx_out = (float*)d_out;                     // [512, 128]
  float* attn    = (float*)d_out + NBOX * DIM_VAL;    // [512, 16384]

  // Workspace layout (bf16 buffers), ~84 MB total
  char* w = (char*)d_ws;
  __bf16* qbf    = (__bf16*)w;                                   // 512*128
  __bf16* keysB  = (__bf16*)(w + 131072);                        // 8*128*16384
  __bf16* valsB  = (__bf16*)(w + 131072 + 33554432ull);          // 8*128*16384
  __bf16* coeffB = (__bf16*)(w + 131072 + 2ull * 33554432ull);   // 512*16384

  qproj_kernel  <<<32,    256, 0, stream>>>(rois, W_proj, b_proj, qbf);
  kv_proj_kernel<<<8192,  256, 0, stream>>>(ctx, W_keys, b_keys, W_vals, b_vals, keysB, valsB);
  qk_kernel     <<<32768, 256, 0, stream>>>(qbf, keysB, bidx, attn);
  softmax_kernel<<<NBOX,  256, 0, stream>>>(attn, coeffB);
  av_kernel     <<<256,   256, 0, stream>>>(coeffB, valsB, bidx, ctx_out);
}
